// EdgePool_30777735643936
// MI455X (gfx1250) — matrix operand order repaired and checked by reading, compiled, not run on hardware
//
#include <hip/hip_runtime.h>
#include <stdint.h>

// ---------------- problem constants (match reference) ----------------
#define NN      20000     // N_NODES
#define NE      200000    // N_EDGES
#define FIN     256
#define HIDC    256
#define NG      128       // N_GRAPHS
#define NC      10        // N_CLASSES
#define NEGBIG  (-1e30f)
#define SORTN   262144    // 2^18 >= NE

typedef __attribute__((ext_vector_type(16))) _Float16 v16h;
typedef __attribute__((ext_vector_type(8)))  _Float16 v8h;
typedef __attribute__((ext_vector_type(8)))  float    v8f;
typedef __attribute__((ext_vector_type(4)))  float    v4f;
typedef unsigned long long u64;
typedef unsigned int       u32;

__device__ __forceinline__ int imin(int a, int b) { return a < b ? a : b; }

// =====================================================================
//  Weight pre-pass: W[K][N] f32 (row-major)  ->  WT[N][K] f16
//  (one-time per launch; weights stay L2-resident afterwards)
// =====================================================================
__global__ void k_wT(const float* __restrict__ W, _Float16* __restrict__ WT,
                     int K, int N) {
    size_t t = (size_t)blockIdx.x * 256 + threadIdx.x;
    if (t >= (size_t)K * N) return;
    int n = (int)(t / K), k = (int)(t % K);
    WT[t] = (_Float16)W[(size_t)k * N + n];
}

// =====================================================================
//  WMMA GEMM:  C[M,N] = A[M,K] @ B[K,N] (+bias)(optional relu)
//  A: f32 row-major.  BT: f16, transposed [N][K].  C: f32 row-major.
//  One 16x16 tile per wave, 8 waves per 256-thread block (16 x 128 of C).
//  K must be a multiple of 32.  Ragged M/N handled by clamped addresses
//  (duplicate compute) + masked stores — the K loop is branch-free.
// =====================================================================
__global__ void k_gemm_wmma(const float* __restrict__ A,
                            const _Float16* __restrict__ BT,
                            const float* __restrict__ bias,
                            float* __restrict__ C,
                            int M, int N, int K, int relu)
{
    const int lane  = threadIdx.x & 31;
    const int wave  = threadIdx.x >> 5;
    const int tileM = blockIdx.y * 16;
    const int tileN = blockIdx.x * 128 + wave * 16;
    const int half  = lane >> 4;       // 0 | 1
    const int l15   = lane & 15;

    const int  rowA  = imin(tileM + l15, M - 1);     // clamped: loop branch-free
    const int  colC  = tileN + l15;
    const bool colOK = colC < N;
    const int  colB  = imin(colC, N - 1);

    const float*    Abase = A  + (size_t)rowA * K + half * 8;   // f32, elems 0..7 at kb
    const _Float16* Bbase = BT + (size_t)colB * K + half * 8;   // f16, K contiguous

    v8f acc = {};
    for (int k0 = 0; k0 < K; k0 += 32) {
        // ---- A fragment (ISA 7.12.2): elems 0..7 = K kb..kb+7,
        //      elems 8..15 = K kb+16..kb+23 ; kb = k0 + half*8 ----
        const v4f* a0p = (const v4f*)(Abase + k0);
        const v4f* a1p = (const v4f*)(Abase + k0 + 16);
        v4f x0 = a0p[0], x1 = a0p[1], x2 = a1p[0], x3 = a1p[1];
        v16h a;
        #pragma unroll
        for (int i = 0; i < 4; ++i) {
            a[i]      = (_Float16)x0[i];
            a[i + 4]  = (_Float16)x1[i];
            a[i + 8]  = (_Float16)x2[i];
            a[i + 12] = (_Float16)x3[i];
        }
        // ---- B fragment: elems 0..7 = K kb..kb+7, elems 8..15 = K kb+16..kb+23 ; kb = k0 + half*8 ----
        const v8h* bp = (const v8h*)(Bbase + k0);
        v8h b0 = bp[0], b1 = bp[2];
        v16h b;
        #pragma unroll
        for (int i = 0; i < 8; ++i) { b[i] = b0[i]; b[i + 8] = b1[i]; }

        acc = __builtin_amdgcn_wmma_f32_16x16x32_f16(
            /*neg_a=*/false, a, /*neg_b=*/false, b,
            /*c_mod=*/(short)0, acc, /*reuse_a=*/false, /*reuse_b=*/false);
    }

    // ---- store: C/D layout — n = lane&15, m = r + 8*half ----
    if (colOK) {
        const float bb = bias ? bias[colC] : 0.0f;
        #pragma unroll
        for (int r = 0; r < 8; ++r) {
            int m = tileM + r + 8 * half;
            if (m < M) {
                float v = acc[r] + bb;
                if (relu) v = fmaxf(v, 0.0f);
                C[(size_t)m * N + colC] = v;
            }
        }
    }
}

// =====================================================================
//  Elementwise / fill / copy helpers
// =====================================================================
__global__ void k_fill_f(float* p, float v, size_t n) {
    size_t t = (size_t)blockIdx.x * 256 + threadIdx.x;
    if (t < n) p[t] = v;
}
__global__ void k_fill_i(int* p, int v, size_t n) {
    size_t t = (size_t)blockIdx.x * 256 + threadIdx.x;
    if (t < n) p[t] = v;
}
__global__ void k_copy_i(int* d, const int* s, size_t n) {
    size_t t = (size_t)blockIdx.x * 256 + threadIdx.x;
    if (t < n) d[t] = s[t];
}

__device__ inline void atomicMaxF(float* addr, float v) {
    u32* ua  = (u32*)addr;
    u32  old = *ua;
    while (true) {
        float f = __uint_as_float(old);
        if (f >= v) break;
        u32 assumed = old;
        old = atomicCAS(ua, assumed, __float_as_uint(v));
        if (old == assumed) break;
    }
}

// =====================================================================
//  GCNConv pieces (symmetric normalization, self-loops weight 1)
// =====================================================================
__global__ void k_deg(const int* src, const int* dst, const int* ev,
                      float* deg, int E) {
    int e = blockIdx.x * 256 + threadIdx.x;
    if (e >= E) return;
    if (ev[e] && src[e] != dst[e]) atomicAdd(&deg[dst[e]], 1.0f);
}
__global__ void k_dinv(const float* deg, float* dinv, int N) {
    int i = blockIdx.x * 256 + threadIdx.x;
    if (i >= N) return;
    dinv[i] = rsqrtf(deg[i] + 1.0f);
}
__global__ void k_self(const float* h, const float* dinv, float* out, int N, int H) {
    size_t t = (size_t)blockIdx.x * 256 + threadIdx.x;
    if (t >= (size_t)N * H) return;
    int i = (int)(t / H);
    float d = dinv[i];
    out[t] = h[t] * d * d;
}
__global__ void k_edge_scatter(const float* h, const int* src, const int* dst,
                               const int* ev, const float* dinv,
                               float* out, int E, int H) {
    size_t t = (size_t)blockIdx.x * 256 + threadIdx.x;
    if (t >= (size_t)E * H) return;
    int e = (int)(t / H), f = (int)(t % H);
    int s = src[e], d = dst[e];
    if (!ev[e] || s == d) return;
    float coef = dinv[s] * dinv[d];
    atomicAdd(&out[(size_t)d * H + f], h[(size_t)s * H + f] * coef);
}
__global__ void k_bias_act(float* out, const float* b, int N, int H, int relu) {
    size_t t = (size_t)blockIdx.x * 256 + threadIdx.x;
    if (t >= (size_t)N * H) return;
    float v = out[t] + b[t % H];
    if (relu) v = fmaxf(v, 0.0f);
    out[t] = v;
}
__global__ void k_concat(const float* a, const float* b, float* cat, int N, int H) {
    size_t t = (size_t)blockIdx.x * 256 + threadIdx.x;
    if (t >= (size_t)N * H) return;
    size_t i = t / H, f = t % H;
    cat[i * (2 * H) + f]     = a[t];
    cat[i * (2 * H) + H + f] = b[t];
}

// =====================================================================
//  Mean pooling by graph (dummy segment NG for masked nodes)
// =====================================================================
__global__ void k_pool_scatter(const float* x, const int* batch, const int* nmask,
                               float* psum, float* pcnt, int N, int H, int G) {
    size_t t = (size_t)blockIdx.x * 256 + threadIdx.x;
    if (t >= (size_t)N * H) return;
    int i = (int)(t / H), f = (int)(t % H);
    int b = nmask[i] ? batch[i] : G;
    if (b < 0 || b > G) b = G;
    atomicAdd(&psum[(size_t)b * H + f], x[t]);
    if (f == 0) atomicAdd(&pcnt[b], 1.0f);
}
__global__ void k_pool_write(const float* psum, const float* pcnt,
                             float* hcat, int slot, int H, int G) {
    size_t t = (size_t)blockIdx.x * 256 + threadIdx.x;
    if (t >= (size_t)G * H) return;
    int g = (int)(t / H), f = (int)(t % H);
    hcat[(size_t)g * 4 * H + slot * H + f] =
        psum[(size_t)g * H + f] / fmaxf(pcnt[g], 1.0f);
}

// =====================================================================
//  EdgePooling
// =====================================================================
__global__ void k_edge_score(const float* x, const int* src, const int* dst,
                             const int* ev, const float* Wp, const float* bp,
                             float* raw, int E, int H) {
    int e = blockIdx.x * 256 + threadIdx.x;
    if (e >= E) return;
    if (!ev[e]) { raw[e] = NEGBIG; return; }
    const float* xs = x + (size_t)src[e] * H;
    const float* xd = x + (size_t)dst[e] * H;
    float acc = bp[0];
    for (int k = 0; k < H; ++k) acc += xs[k] * Wp[k] + xd[k] * Wp[H + k];
    raw[e] = acc;
}
__global__ void k_segmax(const float* raw, const int* dst, const int* ev,
                         float* mx, int E) {
    int e = blockIdx.x * 256 + threadIdx.x;
    if (e >= E) return;
    if (ev[e]) atomicMaxF(&mx[dst[e]], raw[e]);
}
__global__ void k_z_den(float* rawz, const int* dst, const int* ev,
                        const float* mx, float* den, int E) {
    int e = blockIdx.x * 256 + threadIdx.x;
    if (e >= E) return;
    float z = ev[e] ? expf(rawz[e] - mx[dst[e]]) : 0.0f;
    rawz[e] = z;
    if (ev[e]) atomicAdd(&den[dst[e]], z);
}
__global__ void k_e_final(float* ez, const int* dst, const float* den, int E) {
    int e = blockIdx.x * 256 + threadIdx.x;
    if (e >= E) return;
    ez[e] = ez[e] / fmaxf(den[dst[e]], 1e-16f) + 0.5f;
}
__global__ void k_keys_score(const float* es, const int* ev, u64* keys, int E, int SN) {
    int t = blockIdx.x * 256 + threadIdx.x;
    if (t >= SN) return;
    if (t >= E) { keys[t] = ~0ull; return; }
    float kf = ev[t] ? -es[t] : __uint_as_float(0x7F800000u); // +inf
    u32 u = __float_as_uint(kf);
    u = (u & 0x80000000u) ? ~u : (u | 0x80000000u);
    keys[t] = ((u64)u << 32) | (u32)t;
}
__global__ void k_bitonic(u64* keys, int j, int k, int n) {
    int i = blockIdx.x * 256 + threadIdx.x;
    if (i >= n) return;
    int l = i ^ j;
    if (l > i) {
        u64 a = keys[i], b = keys[l];
        bool asc = ((i & k) == 0);
        if ((asc && a > b) || (!asc && a < b)) { keys[i] = b; keys[l] = a; }
    }
}
__global__ void k_init_match(const int* nmask, int* avail, int* cluster, int N) {
    int i = blockIdx.x * 256 + threadIdx.x;
    if (i >= N) return;
    avail[i]   = nmask[i];
    cluster[i] = -1;
}
__global__ void k_greedy(const u64* keys, const int* src, const int* dst,
                         const int* ev, int* avail, int* cluster, int* chosen,
                         int* scal, int E) {
    if (threadIdx.x != 0 || blockIdx.x != 0) return;
    int cnt = 0;
    for (int t = 0; t < E; ++t) {
        int idx = (int)(keys[t] & 0xFFFFFFFFull);
        int s = src[idx], d = dst[idx];
        if (ev[idx] && avail[s] && avail[d]) {
            cluster[s] = cnt; cluster[d] = cnt;
            avail[s] = 0;     avail[d] = 0;
            chosen[idx] = 1;  ++cnt;
        }
    }
    scal[0] = cnt;
}
__global__ void k_finish(const int* nmask, int* avail, int* cluster, int* scal, int N) {
    if (threadIdx.x != 0 || blockIdx.x != 0) return;
    int cnt = scal[0];
    for (int i = 0; i < N; ++i)
        if (avail[i]) cluster[i] = cnt++;
    scal[1] = cnt;                       // n_new
    for (int i = 0; i < N; ++i)
        if (!nmask[i]) cluster[i] = cnt++;
}
__global__ void k_score_set(const int* chosen, const int* cluster, const int* src,
                            const float* es, float* score, int E) {
    int e = blockIdx.x * 256 + threadIdx.x;
    if (e >= E) return;
    if (chosen[e]) score[cluster[src[e]]] = es[e];
}
__global__ void k_newx_scatter(const float* x, const int* cluster, float* nx, int N, int H) {
    size_t t = (size_t)blockIdx.x * 256 + threadIdx.x;
    if (t >= (size_t)N * H) return;
    int i = (int)(t / H), f = (int)(t % H);
    atomicAdd(&nx[(size_t)cluster[i] * H + f], x[t]);
}
__global__ void k_newx_scale(float* nx, const float* score, int N, int H) {
    size_t t = (size_t)blockIdx.x * 256 + threadIdx.x;
    if (t >= (size_t)N * H) return;
    nx[t] *= score[t / H];
}
__global__ void k_keys_edges(const int* cluster, const int* src, const int* dst,
                             const int* ev, u64* keys, int* tmpS, int* tmpD,
                             int E, int N, int SN) {
    int t = blockIdx.x * 256 + threadIdx.x;
    if (t >= SN) return;
    if (t >= E) { keys[t] = ~0ull; return; }
    int ns = cluster[src[t]], nd = cluster[dst[t]];
    tmpS[t] = ns; tmpD[t] = nd;
    u32 key32 = ev[t] ? (u32)((u32)ns * (u32)N + (u32)nd)
                      : (u32)((u32)N * (u32)N + (u32)t);
    keys[t] = ((u64)key32 << 32) | (u32)t;
}
__global__ void k_mark_first(const u64* keys, const int* ev, int* tmpV, int E) {
    int t = blockIdx.x * 256 + threadIdx.x;
    if (t >= E) return;
    u64 kk = keys[t];
    int idx = (int)(kk & 0xFFFFFFFFull);
    int first = (t == 0) || ((keys[t - 1] >> 32) != (kk >> 32));
    tmpV[idx] = first && ev[idx];
}
__global__ void k_commit_edges(int* src, int* dst, int* ev,
                               const int* tS, const int* tD, const int* tV, int E) {
    int e = blockIdx.x * 256 + threadIdx.x;
    if (e >= E) return;
    src[e] = tS[e]; dst[e] = tD[e]; ev[e] = tV[e];
}
__global__ void k_batch_scatter(const int* cluster, const int* batch, int* tB, int N) {
    int i = blockIdx.x * 256 + threadIdx.x;
    if (i >= N) return;
    int c = cluster[i];
    if (c >= 0 && c < N) tB[c] = batch[i];
}
__global__ void k_commit_batch(int* batch, int* nmask, const int* tB,
                               const int* scal, int N) {
    int i = blockIdx.x * 256 + threadIdx.x;
    if (i >= N) return;
    batch[i] = tB[i];
    nmask[i] = (i < scal[1]) ? 1 : 0;
}

__global__ void k_log_softmax(float* out, int G, int C) {
    int g = blockIdx.x * 256 + threadIdx.x;
    if (g >= G) return;
    float* row = out + (size_t)g * C;
    float mx = -3.4e38f;
    for (int i = 0; i < C; ++i) mx = fmaxf(mx, row[i]);
    float s = 0.0f;
    for (int i = 0; i < C; ++i) s += expf(row[i] - mx);
    float lse = mx + logf(s);
    for (int i = 0; i < C; ++i) row[i] -= lse;
}

// =====================================================================
//  Host orchestration
// =====================================================================
static inline unsigned nb(size_t n) { return (unsigned)((n + 255) / 256); }

struct Blk   { const float *W1,*b1,*W2,*b2,*Wl,*bl; };
struct Blk16 { _Float16 *W1,*W2,*Wl; const float *b1,*b2,*bl; };

struct Ctx {
    hipStream_t s;
    float *xA, *xB, *xC, *xD, *xE;            // xE is [NN,512]
    float *psum, *pcnt, *hcat, *hidden;
    float *deg, *dinv, *mx, *den, *score;
    int   *cluster, *avail, *nmask, *batch, *tmpB;
    float *escore;
    int   *src, *dst, *ev, *tmpV, *chosen, *tmpS, *tmpD;
    u64   *keys;
    int   *scal;
};

static void gemm(hipStream_t s, const float* A, const _Float16* BT, const float* bias,
                 float* C, int M, int N, int K, int relu) {
    dim3 g((N + 127) / 128, (M + 15) / 16);
    k_gemm_wmma<<<g, 256, 0, s>>>(A, BT, bias, C, M, N, K, relu);
}

static void bitonic_sort(Ctx& c) {
    for (int k = 2; k <= SORTN; k <<= 1)
        for (int j = k >> 1; j > 0; j >>= 1)
            k_bitonic<<<nb(SORTN), 256, 0, c.s>>>(c.keys, j, k, SORTN);
}

static void gcn_conv(Ctx& c, const float* xin, const _Float16* WT, const float* b,
                     float* out, int relu) {
    gemm(c.s, xin, WT, nullptr, c.xD, NN, HIDC, HIDC, 0);          // h = x @ W
    k_fill_f<<<nb(NN), 256, 0, c.s>>>(c.deg, 0.0f, NN);
    k_deg<<<nb(NE), 256, 0, c.s>>>(c.src, c.dst, c.ev, c.deg, NE);
    k_dinv<<<nb(NN), 256, 0, c.s>>>(c.deg, c.dinv, NN);
    size_t nh = (size_t)NN * HIDC;
    k_self<<<nb(nh), 256, 0, c.s>>>(c.xD, c.dinv, out, NN, HIDC);
    k_edge_scatter<<<nb((size_t)NE * HIDC), 256, 0, c.s>>>(c.xD, c.src, c.dst,
                                                           c.ev, c.dinv, out, NE, HIDC);
    k_bias_act<<<nb(nh), 256, 0, c.s>>>(out, b, NN, HIDC, relu);
}

static void gnn_block(Ctx& c, const float* xin, const Blk16& p, int relu_out) {
    gcn_conv(c, xin, p.W1, p.b1, c.xB, 1);                         // x1
    gcn_conv(c, c.xB, p.W2, p.b2, c.xC, 1);                        // x2
    k_concat<<<nb((size_t)NN * HIDC), 256, 0, c.s>>>(c.xB, c.xC, c.xE, NN, HIDC);
    gemm(c.s, c.xE, p.Wl, p.bl, c.xD, NN, HIDC, 2 * HIDC, relu_out); // out -> xD
}

static void mean_pool(Ctx& c, const float* x, int slot) {
    k_fill_f<<<nb((size_t)(NG + 1) * HIDC), 256, 0, c.s>>>(c.psum, 0.0f, (size_t)(NG + 1) * HIDC);
    k_fill_f<<<nb(NG + 1), 256, 0, c.s>>>(c.pcnt, 0.0f, NG + 1);
    k_pool_scatter<<<nb((size_t)NN * HIDC), 256, 0, c.s>>>(x, c.batch, c.nmask,
                                                           c.psum, c.pcnt, NN, HIDC, NG);
    k_pool_write<<<nb((size_t)NG * HIDC), 256, 0, c.s>>>(c.psum, c.pcnt, c.hcat,
                                                         slot, HIDC, NG);
}

static void edge_pool(Ctx& c, const float* Wp, const float* bp) {
    k_edge_score<<<nb(NE), 256, 0, c.s>>>(c.xD, c.src, c.dst, c.ev, Wp, bp,
                                          c.escore, NE, HIDC);
    k_fill_f<<<nb(NN), 256, 0, c.s>>>(c.mx, NEGBIG, NN);
    k_fill_f<<<nb(NN), 256, 0, c.s>>>(c.den, 0.0f, NN);
    k_segmax<<<nb(NE), 256, 0, c.s>>>(c.escore, c.dst, c.ev, c.mx, NE);
    k_z_den<<<nb(NE), 256, 0, c.s>>>(c.escore, c.dst, c.ev, c.mx, c.den, NE);
    k_e_final<<<nb(NE), 256, 0, c.s>>>(c.escore, c.dst, c.den, NE);
    k_keys_score<<<nb(SORTN), 256, 0, c.s>>>(c.escore, c.ev, c.keys, NE, SORTN);
    bitonic_sort(c);
    k_init_match<<<nb(NN), 256, 0, c.s>>>(c.nmask, c.avail, c.cluster, NN);
    k_fill_i<<<nb(NE), 256, 0, c.s>>>(c.chosen, 0, NE);
    k_greedy<<<1, 1, 0, c.s>>>(c.keys, c.src, c.dst, c.ev, c.avail, c.cluster,
                               c.chosen, c.scal, NE);
    k_finish<<<1, 1, 0, c.s>>>(c.nmask, c.avail, c.cluster, c.scal, NN);
    k_fill_f<<<nb(NN), 256, 0, c.s>>>(c.score, 1.0f, NN);
    k_score_set<<<nb(NE), 256, 0, c.s>>>(c.chosen, c.cluster, c.src, c.escore,
                                         c.score, NE);
    size_t nh = (size_t)NN * HIDC;
    k_fill_f<<<nb(nh), 256, 0, c.s>>>(c.xA, 0.0f, nh);
    k_newx_scatter<<<nb(nh), 256, 0, c.s>>>(c.xD, c.cluster, c.xA, NN, HIDC);
    k_newx_scale<<<nb(nh), 256, 0, c.s>>>(c.xA, c.score, NN, HIDC);
    k_keys_edges<<<nb(SORTN), 256, 0, c.s>>>(c.cluster, c.src, c.dst, c.ev,
                                             c.keys, c.tmpS, c.tmpD, NE, NN, SORTN);
    bitonic_sort(c);
    k_mark_first<<<nb(NE), 256, 0, c.s>>>(c.keys, c.ev, c.tmpV, NE);
    k_commit_edges<<<nb(NE), 256, 0, c.s>>>(c.src, c.dst, c.ev, c.tmpS, c.tmpD,
                                            c.tmpV, NE);
    k_fill_i<<<nb(NN), 256, 0, c.s>>>(c.tmpB, 0, NN);
    k_batch_scatter<<<nb(NN), 256, 0, c.s>>>(c.cluster, c.batch, c.tmpB, NN);
    k_commit_batch<<<nb(NN), 256, 0, c.s>>>(c.batch, c.nmask, c.tmpB, c.scal, NN);
}

extern "C" void kernel_launch(void* const* d_in, const int* in_sizes, int n_in,
                              void* d_out, int out_size, void* d_ws, size_t ws_size,
                              hipStream_t stream) {
    auto F = [&](int i) { return (const float*)d_in[i]; };
    auto I = [&](int i) { return (const int*)d_in[i]; };

    // ---- input mapping (handles insertion-order and jax sorted flatten) ----
    const float* X; const int* EI; const int* BATCH;
    Blk blk[4]; const float *Wp[3], *bp[3];
    const float *FW1, *Fb1, *FW2, *Fb2;
    if (n_in >= 37 && in_sizes[0] == NN * FIN) {
        X = F(0); EI = I(1); BATCH = I(2);
        int t = 3;
        for (int b = 0; b < 4; ++b) { blk[b] = {F(t),F(t+1),F(t+2),F(t+3),F(t+4),F(t+5)}; t += 6; }
        for (int p = 0; p < 3; ++p) { Wp[p] = F(t); bp[p] = F(t+1); t += 2; }
        FW1 = F(t); Fb1 = F(t+1); FW2 = F(t+2); Fb2 = F(t+3);
    } else {
        // sorted: batch, edge_index, W1, W2, b1, b2, blocks{W1,W2,Wl,b1,b2,bl}x4,
        //         pools{Wp,bp}x3, x
        BATCH = I(0); EI = I(1);
        FW1 = F(2); FW2 = F(3); Fb1 = F(4); Fb2 = F(5);
        int t = 6;
        for (int b = 0; b < 4; ++b) { blk[b] = {F(t),F(t+3),F(t+1),F(t+4),F(t+2),F(t+5)}; t += 6; }
        for (int p = 0; p < 3; ++p) { Wp[p] = F(t); bp[p] = F(t+1); t += 2; }
        X = F(t);
    }

    // ---- workspace carve-out (~135 MB) ----
    char* w = (char*)d_ws; size_t off = 0;
    auto A = [&](size_t bytes) -> void* {
        void* p = w + off; off += (bytes + 255) & ~(size_t)255; return p;
    };
    Ctx c; c.s = stream;
    size_t nhB = (size_t)NN * HIDC * sizeof(float);
    c.xA = (float*)A(nhB);  c.xB = (float*)A(nhB);
    c.xC = (float*)A(nhB);  c.xD = (float*)A(nhB);
    c.xE = (float*)A(2 * nhB);
    c.psum   = (float*)A((size_t)(NG + 1) * HIDC * sizeof(float));
    c.pcnt   = (float*)A((NG + 1) * sizeof(float));
    c.hcat   = (float*)A((size_t)NG * 4 * HIDC * sizeof(float));
    c.hidden = (float*)A((size_t)NG * HIDC * sizeof(float));
    c.deg  = (float*)A(NN * 4); c.dinv = (float*)A(NN * 4);
    c.mx   = (float*)A(NN * 4); c.den  = (float*)A(NN * 4);
    c.score= (float*)A(NN * 4);
    c.cluster = (int*)A(NN * 4); c.avail = (int*)A(NN * 4);
    c.nmask   = (int*)A(NN * 4); c.batch = (int*)A(NN * 4);
    c.tmpB    = (int*)A(NN * 4);
    c.escore = (float*)A(NE * 4);
    c.src = (int*)A(NE * 4); c.dst = (int*)A(NE * 4); c.ev = (int*)A(NE * 4);
    c.tmpV = (int*)A(NE * 4); c.chosen = (int*)A(NE * 4);
    c.tmpS = (int*)A(NE * 4); c.tmpD = (int*)A(NE * 4);
    c.keys = (u64*)A((size_t)SORTN * 8);
    c.scal = (int*)A(256);

    // f16 transposed weights (one-time conversion, ~2.6 MB)
    Blk16 b16[4];
    for (int b = 0; b < 4; ++b) {
        b16[b].W1 = (_Float16*)A((size_t)HIDC * HIDC * 2);
        b16[b].W2 = (_Float16*)A((size_t)HIDC * HIDC * 2);
        b16[b].Wl = (_Float16*)A((size_t)2 * HIDC * HIDC * 2);
        b16[b].b1 = blk[b].b1; b16[b].b2 = blk[b].b2; b16[b].bl = blk[b].bl;
    }
    _Float16* FW1T = (_Float16*)A((size_t)4 * HIDC * HIDC * 2);
    _Float16* FW2T = (_Float16*)A((size_t)HIDC * NC * 2);
    (void)ws_size;

    for (int b = 0; b < 4; ++b) {
        k_wT<<<nb((size_t)HIDC * HIDC), 256, 0, stream>>>(blk[b].W1, b16[b].W1, HIDC, HIDC);
        k_wT<<<nb((size_t)HIDC * HIDC), 256, 0, stream>>>(blk[b].W2, b16[b].W2, HIDC, HIDC);
        k_wT<<<nb((size_t)2 * HIDC * HIDC), 256, 0, stream>>>(blk[b].Wl, b16[b].Wl, 2 * HIDC, HIDC);
    }
    k_wT<<<nb((size_t)4 * HIDC * HIDC), 256, 0, stream>>>(FW1, FW1T, 4 * HIDC, HIDC);
    k_wT<<<nb((size_t)HIDC * NC), 256, 0, stream>>>(FW2, FW2T, HIDC, NC);

    // ---- init graph state ----
    k_copy_i<<<nb(NE), 256, 0, stream>>>(c.src, EI, NE);
    k_copy_i<<<nb(NE), 256, 0, stream>>>(c.dst, EI + NE, NE);
    k_fill_i<<<nb(NE), 256, 0, stream>>>(c.ev, 1, NE);
    k_fill_i<<<nb(NN), 256, 0, stream>>>(c.nmask, 1, NN);
    k_copy_i<<<nb(NN), 256, 0, stream>>>(c.batch, BATCH, NN);

    // ---- forward ----
    gnn_block(c, X, b16[0], /*relu=*/1);
    mean_pool(c, c.xD, 0);
    edge_pool(c, Wp[0], bp[0]);                 // new x -> xA
    for (int i = 1; i <= 2; ++i) {
        gnn_block(c, c.xA, b16[i], /*relu=*/1);
        mean_pool(c, c.xD, i);
        edge_pool(c, Wp[i], bp[i]);
    }
    gnn_block(c, c.xA, b16[3], /*relu=*/0);     // embed_final
    mean_pool(c, c.xD, 3);

    // ---- readout MLP + log_softmax ----
    gemm(stream, c.hcat, FW1T, Fb1, c.hidden, NG, HIDC, 4 * HIDC, /*relu=*/1);
    gemm(stream, c.hidden, FW2T, Fb2, (float*)d_out, NG, NC, HIDC, /*relu=*/0);
    k_log_softmax<<<nb(NG), 256, 0, stream>>>((float*)d_out, NG, NC);
    (void)out_size; (void)n_in;
}
